// NetVLAD_46420006535577
// MI455X (gfx1250) — matrix operand order repaired and checked by reading, not compiled, not run on hardware
//
#include <hip/hip_runtime.h>
#include <hip/hip_bf16.h>

// ---------------------------------------------------------------------------
// NetVLAD fused kernel for gfx1250 (MI455X).
//   K1: zero workspace (ax partials + a-sums)
//   K2: fused  [L2-norm(x) -> logits GEMM (WMMA bf16, reg-resident W) ->
//               softmax -> ax GEMM (WMMA bf16) -> global atomic reduction]
//   K3: finalize [ax - asum*centroids, intra-L2, global-L2]
// ---------------------------------------------------------------------------

typedef __attribute__((ext_vector_type(16))) __bf16 v16bf;
typedef __attribute__((ext_vector_type(8)))  float  v8f;

#define B_      16
#define C_      256
#define N_      16384
#define K_      64
#define NCHUNK  32                  // workgroups per batch image
#define PTS_PER_WG (N_/NCHUNK)      // 512 points per WG
#define NT      32                  // points per inner subtile (WMMA k-dim)
#define ITERS   (PTS_PER_WG/NT)     // 16

union Frag {
    unsigned u[8];
    uint4    q[2];
    v16bf    v;
};

__global__ __launch_bounds__(256) void netvlad_zero(float* ws, int n) {
    int i = blockIdx.x * 256 + threadIdx.x;
    if (i < n) ws[i] = 0.f;
}

__global__ __launch_bounds__(256) void netvlad_main(
    const float* __restrict__ x,        // (B,C,N)
    const float* __restrict__ conv_w,   // (K,C)
    const float* __restrict__ conv_b,   // (K)
    float* __restrict__ ax_ws,          // (B,K,C) accum
    float* __restrict__ asum_ws)        // (B,K)   accum
{
    __shared__ float    bias[K_];
    __shared__ __bf16   xnbf[C_ * NT];        // 16 KB  xhat, [c][point], point-major
    __shared__ unsigned xnp [(C_/2) * NT];    // 16 KB  xhat (c,c+1) bf16 pairs per point
    __shared__ float    lg  [K_ * NT];        //  8 KB  logits / exp / asum scratch
    __shared__ __bf16   abf [K_ * NT];        //  4 KB  softmax weights, [k][point]
    __shared__ float    psum[32 * 32];        //  4 KB  reduction scratch
    __shared__ float    rcp0[32];
    __shared__ float    rcp1[32];

    const int tid  = threadIdx.x;
    const int wave = tid >> 5;          // 0..7
    const int lane = tid & 31;
    const int h    = lane >> 4;         // half-wave selector (WMMA layouts)
    const int l16  = lane & 15;
    const int b    = blockIdx.x / NCHUNK;
    const int chunk= blockIdx.x % NCHUNK;
    const int nbase= chunk * PTS_PER_WG;

    const int mt   = wave & 3;          // cluster-row tile (16 rows) owned by wave
    const int half = wave >> 2;         // splits point-tiles / channel-tiles
    const int j    = lane;              // point column (softmax phase)
    const int q    = tid & 7;           // point quad 0..7 (load phase)
    const int cgrp = tid >> 3;          // channel group of 8 rows (load phase)

    if (tid < K_) bias[tid] = conv_b[tid];

    // ---- conv_w A-fragments: loop-invariant, keep resident in registers ----
    Frag aW[8];
    {
        const int arow = mt * 16 + l16;
        #pragma unroll
        for (int s = 0; s < 8; ++s) {
            const int c0 = s * 32;
            #pragma unroll
            for (int e = 0; e < 8; ++e) {
                aW[s].v[e]     = (__bf16)conv_w[arow * C_ + c0 + h * 8 + e];
                aW[s].v[8 + e] = (__bf16)conv_w[arow * C_ + c0 + 16 + h * 8 + e];
            }
        }
    }

    // persistent ax accumulators: wave owns (mt, ct = half*8 + t) 16x16 tiles
    v8f acc[8];
    #pragma unroll
    for (int t = 0; t < 8; ++t) acc[t] = (v8f){0,0,0,0,0,0,0,0};
    float asum_part[8];
    #pragma unroll
    for (int r = 0; r < 8; ++r) asum_part[r] = 0.f;

    __syncthreads();

    for (int it = 0; it < ITERS; ++it) {
        const int n0 = nbase + it * NT;

        // ==== 1. load x[b,:,n0:n0+32] as float4, per-point L2 norm -> bf16 ==
        const float* xbase = x + ((size_t)b * C_ + cgrp * 8) * N_ + n0 + 4 * q;
        float v[8][4];
        float ps[4] = {0.f, 0.f, 0.f, 0.f};
        #pragma unroll
        for (int r = 0; r < 8; ++r) {          // 8 x global_load_b128 per thread
            float4 f = *(const float4*)(xbase + (size_t)r * N_);
            v[r][0] = f.x; v[r][1] = f.y; v[r][2] = f.z; v[r][3] = f.w;
            ps[0] += f.x * f.x; ps[1] += f.y * f.y;
            ps[2] += f.z * f.z; ps[3] += f.w * f.w;
        }
        if (it + 1 < ITERS) __builtin_prefetch(xbase + NT, 0, 1);
        #pragma unroll
        for (int i = 0; i < 4; ++i) psum[cgrp * 32 + 4 * q + i] = ps[i];
        __syncthreads();
        if (tid < 32) {
            float s = 0.f;
            #pragma unroll
            for (int g = 0; g < 32; ++g) s += psum[g * 32 + tid];
            rcp0[tid] = 1.0f / fmaxf(sqrtf(s), 1e-12f);
        }
        __syncthreads();
        {
            float sc[4];
            #pragma unroll
            for (int i = 0; i < 4; ++i) sc[i] = rcp0[4 * q + i];
            unsigned short bits[8][4];
            #pragma unroll
            for (int r = 0; r < 8; ++r)
                #pragma unroll
                for (int i = 0; i < 4; ++i)
                    bits[r][i] = __builtin_bit_cast(unsigned short,
                                                    (__bf16)(v[r][i] * sc[i]));
            // xnbf[c][point]: packed point-pairs -> 2 dword stores per row
            unsigned* xn32 = (unsigned*)xnbf;
            #pragma unroll
            for (int r = 0; r < 8; ++r) {
                const int c = cgrp * 8 + r;
                xn32[c * (NT/2) + 2 * q]     = (unsigned)bits[r][0] | ((unsigned)bits[r][1] << 16);
                xn32[c * (NT/2) + 2 * q + 1] = (unsigned)bits[r][2] | ((unsigned)bits[r][3] << 16);
            }
            // xnp[(c/2)][point]: (c,c+1) channel-pairs per point
            #pragma unroll
            for (int rp = 0; rp < 4; ++rp) {
                const int cp = cgrp * 4 + rp;  // channel pair index
                #pragma unroll
                for (int i = 0; i < 4; ++i)
                    xnp[cp * NT + 4 * q + i] =
                        (unsigned)bits[2 * rp][i] | ((unsigned)bits[2 * rp + 1][i] << 16);
            }
        }
        __syncthreads();

        // ==== 2. logits(64x32) = conv_w(64x256) * xhat(256x32)  via WMMA ====
        const int jcol = half * 16 + l16;
        v8f lacc = (v8f){0,0,0,0,0,0,0,0};
        #pragma unroll
        for (int s = 0; s < 8; ++s) {
            Frag Bf;
            #pragma unroll
            for (int p = 0; p < 8; ++p)        // 8 aligned ds_load_b32
                Bf.u[p] = xnp[(s * 16 + (p & 3) + (p >> 2) * 8 + h * 4) * NT + jcol];
            lacc = __builtin_amdgcn_wmma_f32_16x16x32_bf16(
                       false, aW[s].v, false, Bf.v, (short)0, lacc, false, false);
        }
        #pragma unroll
        for (int r = 0; r < 8; ++r) {          // D layout: M = r + 8*h
            const int row = mt * 16 + h * 8 + r;
            lg[row * NT + jcol] = lacc[r] + bias[row];
        }
        __syncthreads();

        // ==== 3. softmax over K=64 per point (cross-wave via LDS) ===========
        {
            float mx = -3.0e38f;
            #pragma unroll
            for (int r = 0; r < 8; ++r) mx = fmaxf(mx, lg[(wave * 8 + r) * NT + j]);
            psum[wave * 32 + j] = mx;
        }
        __syncthreads();
        if (tid < 32) {
            float m = -3.0e38f;
            #pragma unroll
            for (int g = 0; g < 8; ++g) m = fmaxf(m, psum[g * 32 + tid]);
            rcp0[tid] = m;
        }
        __syncthreads();
        {
            const float pm = rcp0[j];
            float se = 0.f;
            #pragma unroll
            for (int r = 0; r < 8; ++r) {
                const int row = wave * 8 + r;
                float e = __expf(lg[row * NT + j] - pm);
                lg[row * NT + j] = e;
                se += e;
            }
            psum[wave * 32 + j] = se;
        }
        __syncthreads();
        if (tid < 32) {
            float s = 0.f;
            #pragma unroll
            for (int g = 0; g < 8; ++g) s += psum[g * 32 + tid];
            rcp1[tid] = 1.0f / s;
        }
        __syncthreads();
        {
            const float inv = rcp1[j];
            #pragma unroll
            for (int r = 0; r < 8; ++r) {
                const int row = wave * 8 + r;
                float a = lg[row * NT + j] * inv;
                abf[row * NT + j] = (__bf16)a;    // A operand for ax GEMM
                asum_part[r] += a;               // per-thread a row-sum partial
            }
        }
        __syncthreads();

        // ==== 4. ax(64x256) += a(64x32) * xhat^T(32x256)  via WMMA ==========
        {
            Frag Af;
            const int arow = mt * 16 + l16;
            Af.q[0] = *(const uint4*)&abf[arow * NT + h * 8];
            Af.q[1] = *(const uint4*)&abf[arow * NT + 16 + h * 8];
            #pragma unroll
            for (int t = 0; t < 8; ++t) {
                const int ccol = (half * 8 + t) * 16 + l16;   // channel column
                Frag Bf;
                #pragma unroll
                for (int p = 0; p < 8; ++p)                   // point pairs contiguous
                    Bf.u[p] = *(const unsigned*)&xnbf[ccol * NT + 2 * p + (p >> 2) * 8 + h * 8];
                acc[t] = __builtin_amdgcn_wmma_f32_16x16x32_bf16(
                             false, Af.v, false, Bf.v, (short)0, acc[t], false, false);
            }
        }
        // next iteration's first barrier orders these reads vs. LDS rewrites
    }

    // ==== epilogue: reduce partials to global workspace =====================
    #pragma unroll
    for (int t = 0; t < 8; ++t) {
        const int ct = half * 8 + t;
        #pragma unroll
        for (int r = 0; r < 8; ++r) {
            const int row = mt * 16 + h * 8 + r;
            const int col = ct * 16 + l16;
            atomicAdd(&ax_ws[((size_t)b * K_ + row) * C_ + col], acc[t][r]);
        }
    }
    #pragma unroll
    for (int r = 0; r < 8; ++r) lg[(wave * 8 + r) * NT + j] = asum_part[r];
    __syncthreads();
    if (tid < K_) {
        float s = 0.f;
        #pragma unroll
        for (int p = 0; p < NT; ++p) s += lg[tid * NT + p];
        atomicAdd(&asum_ws[b * K_ + tid], s);
    }
}

__global__ __launch_bounds__(256) void netvlad_finalize(
    const float* __restrict__ ax_ws, const float* __restrict__ asum_ws,
    const float* __restrict__ centroids, float* __restrict__ out)
{
    __shared__ float red[256];
    __shared__ float gnorm;
    const int b = blockIdx.x;
    const int c = threadIdx.x;
    float gss = 0.f;
    for (int k = 0; k < K_; ++k) {
        float v = ax_ws[((size_t)b * K_ + k) * C_ + c]
                - asum_ws[b * K_ + k] * centroids[k * C_ + c];
        red[c] = v * v;
        __syncthreads();
        for (int off = 128; off > 0; off >>= 1) {
            if (c < off) red[c] += red[c + off];
            __syncthreads();
        }
        float iv = 1.0f / fmaxf(sqrtf(red[0]), 1e-12f);
        __syncthreads();
        float vn = v * iv;
        out[(size_t)b * (K_ * C_) + k * C_ + c] = vn;
        gss += vn * vn;
    }
    red[c] = gss;
    __syncthreads();
    for (int off = 128; off > 0; off >>= 1) {
        if (c < off) red[c] += red[c + off];
        __syncthreads();
    }
    if (c == 0) gnorm = 1.0f / fmaxf(sqrtf(red[0]), 1e-12f);
    __syncthreads();
    const float gi = gnorm;
    for (int k = 0; k < K_; ++k)
        out[(size_t)b * (K_ * C_) + k * C_ + c] *= gi;
}

extern "C" void kernel_launch(void* const* d_in, const int* in_sizes, int n_in,
                              void* d_out, int out_size, void* d_ws, size_t ws_size,
                              hipStream_t stream) {
    const float* x         = (const float*)d_in[0];   // (16,256,16384)
    const float* conv_w    = (const float*)d_in[1];   // (64,256)
    const float* conv_b    = (const float*)d_in[2];   // (64)
    const float* centroids = (const float*)d_in[3];   // (64,256)
    float* out   = (float*)d_out;                     // (16, 16384)
    float* ax_ws = (float*)d_ws;                      // B*K*C floats
    float* asum_ws = ax_ws + (size_t)B_ * K_ * C_;    // B*K floats

    const int zn = B_ * K_ * C_ + B_ * K_;
    netvlad_zero<<<(zn + 255) / 256, 256, 0, stream>>>((float*)d_ws, zn);
    netvlad_main<<<B_ * NCHUNK, 256, 0, stream>>>(x, conv_w, conv_b, ax_ws, asum_ws);
    netvlad_finalize<<<B_, 256, 0, stream>>>(ax_ws, asum_ws, centroids, out);
}
